// CHGNet_6648609374619
// MI455X (gfx1250) — matrix operand order repaired and checked by reading, not compiled, not run on hardware
//
#include <hip/hip_runtime.h>
#include <hip/hip_bf16.h>

// ---------------------------------------------------------------------------
// CHGNet forward, fused for MI455X (gfx1250): bf16 WMMA 16x16x32, f32 accum.
// ---------------------------------------------------------------------------

typedef __bf16 bf16_t;
typedef bf16_t bf16x8  __attribute__((ext_vector_type(8)));
typedef bf16_t bf16x16 __attribute__((ext_vector_type(16)));
typedef float  f32x4   __attribute__((ext_vector_type(4)));
typedef float  f32x8   __attribute__((ext_vector_type(8)));

#define N_ATOMS  8192
#define N_UND    65536
#define N_DIR    131072
#define N_ANG    262144
#define N_STRUCT 128

__device__ __forceinline__ f32x8 wmma_bf16(bf16x16 a, bf16x16 b, f32x8 c) {
  return __builtin_amdgcn_wmma_f32_16x16x32_bf16(false, a, false, b, (short)0, c,
                                                 false, false);
}

// A fragment (16x32 bf16, M x K) from an LDS row-major bf16 tile.
// Lane layout (ISA 7.12.2): lanes 0-15 hold K=[0..8)+[16..24), lanes 16-31 hold
// K=[8..16)+[24..32) of row M=lane&15 -> two contiguous 16B LDS loads per lane.
__device__ __forceinline__ bf16x16 lds_a_frag(const bf16_t* __restrict__ s, int m,
                                              int k0, int stride, int koff) {
  const bf16_t* p = s + m * stride + k0 + koff;
  bf16x8 lo = *(const bf16x8*)(p);
  bf16x8 hi = *(const bf16x8*)(p + 16);
  return __builtin_shufflevector(lo, hi, 0, 1, 2, 3, 4, 5, 6, 7, 8, 9, 10, 11, 12,
                                 13, 14, 15);
}

// B fragment (32x16 bf16, K x N) from globally pre-packed fragment order:
// contiguous 32 bytes per lane per tile.
__device__ __forceinline__ bf16x16 glb_b_frag(const bf16_t* __restrict__ p, int tile,
                                              int lane) {
  return *(const bf16x16*)(p + (((long)tile << 5) + lane) * 16);
}

__device__ __forceinline__ float sig_(float x)  { return 1.f / (1.f + __expf(-x)); }
__device__ __forceinline__ float silu_(float x) { return x * sig_(x); }

// ---------------------------------------------------------------------------
// Weight repack: [K][64] f32 row-major -> WMMA-B fragment order bf16.
// dst[((kt*4+nt)*32 + lane)*16 + i] = W[kt*32 + (i>>3)*16 + (lane>>4)*8 + (i&7)][nt*16 + (lane&15)]
// ---------------------------------------------------------------------------
__global__ void k_pack_w(const float* __restrict__ src, bf16_t* __restrict__ dst,
                         int kn) {
  int t = blockIdx.x * 256 + threadIdx.x;
  if (t >= kn) return;
  int k = t >> 6, n = t & 63;
  int kt = k >> 5, rem = k & 31;
  int l = (((rem >> 3) & 1) << 4) | (n & 15);
  int i = (rem & 7) | ((rem >> 4) << 3);
  int tile = kt * 4 + (n >> 4);
  dst[(tile * 32 + l) * 16 + i] = (bf16_t)src[t];
}

// ---------------------------------------------------------------------------
// Embedding stage (K=9 GEMMs -> plain VALU; WMMA cannot help at K=9)
// ---------------------------------------------------------------------------
__global__ void k_atom_embed(const int* __restrict__ an, const float* __restrict__ emb,
                             float* __restrict__ atom) {
  int t = blockIdx.x * 256 + threadIdx.x;  // N_ATOMS*64
  int row = t >> 6, c = t & 63;
  atom[t] = emb[an[row] * 64 + c];
}

__global__ void k_bond_embed(const float* __restrict__ bbag,
                             const float* __restrict__ bbbg,
                             const float* __restrict__ Wb,
                             const float* __restrict__ Wwa,
                             const float* __restrict__ Wwb, float* __restrict__ bond,
                             float* __restrict__ bwag, float* __restrict__ bwbg) {
  int t = blockIdx.x * 256 + threadIdx.x;  // N_UND*64
  int row = t >> 6, c = t & 63;
  float sb = 0.f, sa = 0.f, sg = 0.f;
#pragma unroll
  for (int k = 0; k < 9; ++k) {
    float a = bbag[row * 9 + k];
    float g = bbbg[row * 9 + k];
    sb += a * Wb[k * 64 + c];
    sa += a * Wwa[k * 64 + c];
    sg += g * Wwb[k * 64 + c];
  }
  bond[t] = sb;
  bwag[t] = sa;
  bwbg[t] = sg;
}

__global__ void k_angle_embed(const float* __restrict__ ab, const float* __restrict__ Wa,
                              float* __restrict__ angle) {
  int t = blockIdx.x * 256 + threadIdx.x;  // N_ANG*64
  int row = t >> 6, c = t & 63;
  float s = 0.f;
#pragma unroll
  for (int k = 0; k < 9; ++k) s += ab[row * 9 + k] * Wa[k * 64 + c];
  angle[t] = s;
}

// ---------------------------------------------------------------------------
// AtomConv message kernel: 16 edges/block. concat[atom[ag0],bond[d2u],atom[ag1]]
// (K=192) -> gated MLP (192->64->64) -> *bw_ag[d2u] -> atomic scatter to ag0.
// ---------------------------------------------------------------------------
__global__ __launch_bounds__(128) void k_atom_msg(
    const float* __restrict__ atom, const float* __restrict__ bond,
    const float* __restrict__ bw_ag, const int* __restrict__ ag,
    const int* __restrict__ d2u, const bf16_t* __restrict__ cw1p,
    const float* __restrict__ cb1, const bf16_t* __restrict__ gw1p,
    const float* __restrict__ gb1, const bf16_t* __restrict__ cw2p,
    const float* __restrict__ cb2, const bf16_t* __restrict__ gw2p,
    const float* __restrict__ gb2, float* __restrict__ atom_acc) {
  __shared__ alignas(32) bf16_t sA[16 * 192];
  __shared__ alignas(32) bf16_t sH[2][16 * 64];
  __shared__ int sDst[16];
  __shared__ int sBw[16];

  const int tid = threadIdx.x;
  const int e0 = blockIdx.x * 16;
  {  // gather + f32->bf16 staging: thread (r,seg) loads 32 features of row r
    int r = tid >> 3, seg = tid & 7;
    if (seg < 6) {
      int e = e0 + r;
      int i0 = ag[2 * e], i1 = ag[2 * e + 1], u = d2u[e];
      if (seg == 0) { sDst[r] = i0; sBw[r] = u; }
      const float* src;
      if (seg < 2)      src = atom + i0 * 64 + seg * 32;
      else if (seg < 4) src = bond + u * 64 + (seg - 2) * 32;
      else              src = atom + i1 * 64 + (seg - 4) * 32;
      bf16_t* dst = sA + r * 192 + seg * 32;
#pragma unroll
      for (int j = 0; j < 8; ++j) {
        f32x4 v = *(const f32x4*)(src + j * 4);
        dst[j * 4 + 0] = (bf16_t)v.x; dst[j * 4 + 1] = (bf16_t)v.y;
        dst[j * 4 + 2] = (bf16_t)v.z; dst[j * 4 + 3] = (bf16_t)v.w;
      }
    }
  }
  __syncthreads();

  const int lane = tid & 31, nt = tid >> 5;      // wave nt owns 16 output cols
  const int hl = lane >> 4, nlo = lane & 15;
  const int col = nt * 16 + nlo, koff = hl * 8;

  f32x8 c0 = {}, c1 = {};
#pragma unroll
  for (int ks = 0; ks < 6; ++ks) {               // K = 192
    bf16x16 a = lds_a_frag(sA, nlo, ks * 32, 192, koff);
    c0 = wmma_bf16(a, glb_b_frag(cw1p, ks * 4 + nt, lane), c0);
    c1 = wmma_bf16(a, glb_b_frag(gw1p, ks * 4 + nt, lane), c1);
  }
  float b1c = cb1[col], b1g = gb1[col];
#pragma unroll
  for (int r = 0; r < 8; ++r) {
    int m = r + hl * 8;
    sH[0][m * 64 + col] = (bf16_t)silu_(c0[r] + b1c);
    sH[1][m * 64 + col] = (bf16_t)silu_(c1[r] + b1g);
  }
  __syncthreads();

  f32x8 d0 = {}, d1 = {};
#pragma unroll
  for (int ks = 0; ks < 2; ++ks) {               // K = 64
    bf16x16 ac = lds_a_frag(sH[0], nlo, ks * 32, 64, koff);
    bf16x16 agf = lds_a_frag(sH[1], nlo, ks * 32, 64, koff);
    d0 = wmma_bf16(ac, glb_b_frag(cw2p, ks * 4 + nt, lane), d0);
    d1 = wmma_bf16(agf, glb_b_frag(gw2p, ks * 4 + nt, lane), d1);
  }
  float b2c = cb2[col], b2g = gb2[col];
#pragma unroll
  for (int r = 0; r < 8; ++r) {
    int m = r + hl * 8;
    float v = silu_(d0[r] + b2c) * sig_(d1[r] + b2g) * bw_ag[sBw[m] * 64 + col];
    atomicAdd(atom_acc + sDst[m] * 64 + col, v);
  }
}

// ---------------------------------------------------------------------------
// BondConv message kernel: 16 angle-rows/block, K=256 concat, scatter to bgi.
// ---------------------------------------------------------------------------
__global__ __launch_bounds__(128) void k_bond_msg(
    const float* __restrict__ atom, const float* __restrict__ bond,
    const float* __restrict__ angle, const float* __restrict__ bw_bg,
    const int* __restrict__ bga, const int* __restrict__ bgb,
    const bf16_t* __restrict__ cw1p, const float* __restrict__ cb1,
    const bf16_t* __restrict__ gw1p, const float* __restrict__ gb1,
    const bf16_t* __restrict__ cw2p, const float* __restrict__ cb2,
    const bf16_t* __restrict__ gw2p, const float* __restrict__ gb2,
    float* __restrict__ bond_acc) {
  __shared__ alignas(32) bf16_t sA[16 * 256];
  __shared__ alignas(32) bf16_t sH[2][16 * 64];
  __shared__ int sDst[16];

  const int tid = threadIdx.x;
  const int e0 = blockIdx.x * 16;
  {
    int r = tid >> 3, seg = tid & 7;
    int e = e0 + r;
    int bi = bgb[2 * e], bj = bgb[2 * e + 1];
    if (seg == 0) sDst[r] = bi;
    const float* src;
    if (seg < 2)      src = bond + bi * 64 + seg * 32;
    else if (seg < 4) src = bond + bj * 64 + (seg - 2) * 32;
    else if (seg < 6) src = angle + e * 64 + (seg - 4) * 32;
    else              src = atom + bga[e] * 64 + (seg - 6) * 32;
    bf16_t* dst = sA + r * 256 + seg * 32;
#pragma unroll
    for (int j = 0; j < 8; ++j) {
      f32x4 v = *(const f32x4*)(src + j * 4);
      dst[j * 4 + 0] = (bf16_t)v.x; dst[j * 4 + 1] = (bf16_t)v.y;
      dst[j * 4 + 2] = (bf16_t)v.z; dst[j * 4 + 3] = (bf16_t)v.w;
    }
  }
  __syncthreads();

  const int lane = tid & 31, nt = tid >> 5;
  const int hl = lane >> 4, nlo = lane & 15;
  const int col = nt * 16 + nlo, koff = hl * 8;

  f32x8 c0 = {}, c1 = {};
#pragma unroll
  for (int ks = 0; ks < 8; ++ks) {               // K = 256
    bf16x16 a = lds_a_frag(sA, nlo, ks * 32, 256, koff);
    c0 = wmma_bf16(a, glb_b_frag(cw1p, ks * 4 + nt, lane), c0);
    c1 = wmma_bf16(a, glb_b_frag(gw1p, ks * 4 + nt, lane), c1);
  }
  float b1c = cb1[col], b1g = gb1[col];
#pragma unroll
  for (int r = 0; r < 8; ++r) {
    int m = r + hl * 8;
    sH[0][m * 64 + col] = (bf16_t)silu_(c0[r] + b1c);
    sH[1][m * 64 + col] = (bf16_t)silu_(c1[r] + b1g);
  }
  __syncthreads();

  f32x8 d0 = {}, d1 = {};
#pragma unroll
  for (int ks = 0; ks < 2; ++ks) {
    bf16x16 ac = lds_a_frag(sH[0], nlo, ks * 32, 64, koff);
    bf16x16 agf = lds_a_frag(sH[1], nlo, ks * 32, 64, koff);
    d0 = wmma_bf16(ac, glb_b_frag(cw2p, ks * 4 + nt, lane), d0);
    d1 = wmma_bf16(agf, glb_b_frag(gw2p, ks * 4 + nt, lane), d1);
  }
  float b2c = cb2[col], b2g = gb2[col];
#pragma unroll
  for (int r = 0; r < 8; ++r) {
    int m = r + hl * 8;
    float v = silu_(d0[r] + b2c) * sig_(d1[r] + b2g) * bw_bg[sDst[m] * 64 + col];
    atomicAdd(bond_acc + sDst[m] * 64 + col, v);
  }
}

// ---------------------------------------------------------------------------
// AngleUpdate: same K=256 concat (with updated bond), single gated layer,
// residual in place (each block reads/writes only its own angle rows).
// ---------------------------------------------------------------------------
__global__ __launch_bounds__(128) void k_angle_upd(
    const float* __restrict__ atom, const float* __restrict__ bond,
    float* __restrict__ angle, const int* __restrict__ bga,
    const int* __restrict__ bgb, const bf16_t* __restrict__ cwp,
    const float* __restrict__ cb, const bf16_t* __restrict__ gwp,
    const float* __restrict__ gb) {
  __shared__ alignas(32) bf16_t sA[16 * 256];
  const int tid = threadIdx.x;
  const int e0 = blockIdx.x * 16;
  {
    int r = tid >> 3, seg = tid & 7;
    int e = e0 + r;
    int bi = bgb[2 * e], bj = bgb[2 * e + 1];
    const float* src;
    if (seg < 2)      src = bond + bi * 64 + seg * 32;
    else if (seg < 4) src = bond + bj * 64 + (seg - 2) * 32;
    else if (seg < 6) src = angle + e * 64 + (seg - 4) * 32;
    else              src = atom + bga[e] * 64 + (seg - 6) * 32;
    bf16_t* dst = sA + r * 256 + seg * 32;
#pragma unroll
    for (int j = 0; j < 8; ++j) {
      f32x4 v = *(const f32x4*)(src + j * 4);
      dst[j * 4 + 0] = (bf16_t)v.x; dst[j * 4 + 1] = (bf16_t)v.y;
      dst[j * 4 + 2] = (bf16_t)v.z; dst[j * 4 + 3] = (bf16_t)v.w;
    }
  }
  __syncthreads();

  const int lane = tid & 31, nt = tid >> 5;
  const int hl = lane >> 4, nlo = lane & 15;
  const int col = nt * 16 + nlo, koff = hl * 8;

  f32x8 c0 = {}, c1 = {};
#pragma unroll
  for (int ks = 0; ks < 8; ++ks) {
    bf16x16 a = lds_a_frag(sA, nlo, ks * 32, 256, koff);
    c0 = wmma_bf16(a, glb_b_frag(cwp, ks * 4 + nt, lane), c0);
    c1 = wmma_bf16(a, glb_b_frag(gwp, ks * 4 + nt, lane), c1);
  }
  float bc = cb[col], bg = gb[col];
#pragma unroll
  for (int r = 0; r < 8; ++r) {
    int m = r + hl * 8;
    angle[(e0 + m) * 64 + col] += silu_(c0[r] + bc) * sig_(c1[r] + bg);
  }
}

// ---------------------------------------------------------------------------
// dst[16 rows] += acc[16 rows] @ W (64x64, packed), residual out-projection.
// ---------------------------------------------------------------------------
__global__ __launch_bounds__(128) void k_mat_out(const float* __restrict__ acc,
                                                 const bf16_t* __restrict__ Wp,
                                                 float* __restrict__ dst) {
  __shared__ alignas(32) bf16_t sA[16 * 64];
  const int tid = threadIdx.x;
  const int base = blockIdx.x * 1024;
  {
    const float* s = acc + base + tid * 8;
    f32x4 v0 = *(const f32x4*)(s);
    f32x4 v1 = *(const f32x4*)(s + 4);
    bf16_t* d = sA + tid * 8;
    d[0] = (bf16_t)v0.x; d[1] = (bf16_t)v0.y; d[2] = (bf16_t)v0.z; d[3] = (bf16_t)v0.w;
    d[4] = (bf16_t)v1.x; d[5] = (bf16_t)v1.y; d[6] = (bf16_t)v1.z; d[7] = (bf16_t)v1.w;
  }
  __syncthreads();
  const int lane = tid & 31, nt = tid >> 5;
  const int hl = lane >> 4, nlo = lane & 15;
  const int col = nt * 16 + nlo, koff = hl * 8;
  f32x8 c = {};
#pragma unroll
  for (int ks = 0; ks < 2; ++ks)
    c = wmma_bf16(lds_a_frag(sA, nlo, ks * 32, 64, koff),
                  glb_b_frag(Wp, ks * 4 + nt, lane), c);
#pragma unroll
  for (int r = 0; r < 8; ++r) {
    int m = r + hl * 8;
    dst[base + m * 64 + col] += c[r];
  }
}

// ---------------------------------------------------------------------------
// Readout: silu(atom@W1+b1) -> silu(@W2+b2) -> @w3+b3 -> segment sums.
// ---------------------------------------------------------------------------
__global__ __launch_bounds__(128) void k_readout(
    const float* __restrict__ atom, const bf16_t* __restrict__ w1p,
    const float* __restrict__ b1, const bf16_t* __restrict__ w2p,
    const float* __restrict__ b2, const float* __restrict__ w3,
    const float* __restrict__ b3, const int* __restrict__ owners,
    float* __restrict__ esum, float* __restrict__ cnt) {
  __shared__ alignas(32) bf16_t sA[16 * 64];
  __shared__ alignas(32) bf16_t sH[16 * 64];
  __shared__ alignas(32) bf16_t sG[16 * 64];
  const int tid = threadIdx.x;
  const int base = blockIdx.x * 1024;
  {
    const float* s = atom + base + tid * 8;
    f32x4 v0 = *(const f32x4*)(s);
    f32x4 v1 = *(const f32x4*)(s + 4);
    bf16_t* d = sA + tid * 8;
    d[0] = (bf16_t)v0.x; d[1] = (bf16_t)v0.y; d[2] = (bf16_t)v0.z; d[3] = (bf16_t)v0.w;
    d[4] = (bf16_t)v1.x; d[5] = (bf16_t)v1.y; d[6] = (bf16_t)v1.z; d[7] = (bf16_t)v1.w;
  }
  __syncthreads();
  const int lane = tid & 31, nt = tid >> 5;
  const int hl = lane >> 4, nlo = lane & 15;
  const int col = nt * 16 + nlo, koff = hl * 8;

  f32x8 c = {};
#pragma unroll
  for (int ks = 0; ks < 2; ++ks)
    c = wmma_bf16(lds_a_frag(sA, nlo, ks * 32, 64, koff),
                  glb_b_frag(w1p, ks * 4 + nt, lane), c);
  float bb1 = b1[col];
#pragma unroll
  for (int r = 0; r < 8; ++r) sH[(r + hl * 8) * 64 + col] = (bf16_t)silu_(c[r] + bb1);
  __syncthreads();

  f32x8 d = {};
#pragma unroll
  for (int ks = 0; ks < 2; ++ks)
    d = wmma_bf16(lds_a_frag(sH, nlo, ks * 32, 64, koff),
                  glb_b_frag(w2p, ks * 4 + nt, lane), d);
  float bb2 = b2[col];
#pragma unroll
  for (int r = 0; r < 8; ++r) sG[(r + hl * 8) * 64 + col] = (bf16_t)silu_(d[r] + bb2);
  __syncthreads();

  if (tid < 16) {
    float e = b3[0];
#pragma unroll
    for (int k = 0; k < 64; ++k) e += (float)sG[tid * 64 + k] * w3[k];
    int o = owners[blockIdx.x * 16 + tid];
    atomicAdd(esum + o, e);
    atomicAdd(cnt + o, 1.f);
  }
}

__global__ void k_finalize(const float* __restrict__ esum, const float* __restrict__ cnt,
                           float* __restrict__ out) {
  int s = threadIdx.x;
  if (s < N_STRUCT) out[s] = esum[s] / fmaxf(cnt[s], 1.f);
}

// ---------------------------------------------------------------------------
extern "C" void kernel_launch(void* const* d_in, const int* in_sizes, int n_in,
                              void* d_out, int out_size, void* d_ws, size_t ws_size,
                              hipStream_t stream) {
  (void)in_sizes; (void)n_in; (void)out_size; (void)ws_size;
  const int*   atomic_numbers = (const int*)d_in[0];
  const float* bb_ag  = (const float*)d_in[1];
  const float* bb_bg  = (const float*)d_in[2];
  const float* ang_b  = (const float*)d_in[3];
  const int*   ag     = (const int*)d_in[4];
  const int*   d2u    = (const int*)d_in[5];
  const int*   bga    = (const int*)d_in[6];
  const int*   bgb    = (const int*)d_in[7];
  const int*   owners = (const int*)d_in[8];
  const float* emb    = (const float*)d_in[9];
  const float* Wb     = (const float*)d_in[10];
  const float* Wa     = (const float*)d_in[11];
  const float* Wbw_ag = (const float*)d_in[12];
  const float* Wbw_bg = (const float*)d_in[13];
  const float* ac_cw1 = (const float*)d_in[14];
  const float* ac_cb1 = (const float*)d_in[15];
  const float* ac_cw2 = (const float*)d_in[16];
  const float* ac_cb2 = (const float*)d_in[17];
  const float* ac_gw1 = (const float*)d_in[18];
  const float* ac_gb1 = (const float*)d_in[19];
  const float* ac_gw2 = (const float*)d_in[20];
  const float* ac_gb2 = (const float*)d_in[21];
  const float* ac_out = (const float*)d_in[22];
  const float* bc_cw1 = (const float*)d_in[23];
  const float* bc_cb1 = (const float*)d_in[24];
  const float* bc_cw2 = (const float*)d_in[25];
  const float* bc_cb2 = (const float*)d_in[26];
  const float* bc_gw1 = (const float*)d_in[27];
  const float* bc_gb1 = (const float*)d_in[28];
  const float* bc_gw2 = (const float*)d_in[29];
  const float* bc_gb2 = (const float*)d_in[30];
  const float* bc_out = (const float*)d_in[31];
  const float* an_cw  = (const float*)d_in[32];
  const float* an_cb  = (const float*)d_in[33];
  const float* an_gw  = (const float*)d_in[34];
  const float* an_gb  = (const float*)d_in[35];
  const float* ro_w1  = (const float*)d_in[36];
  const float* ro_b1  = (const float*)d_in[37];
  const float* ro_w2  = (const float*)d_in[38];
  const float* ro_b2  = (const float*)d_in[39];
  const float* ro_w3  = (const float*)d_in[40];
  const float* ro_b3  = (const float*)d_in[41];

  // ---- workspace carve (256B aligned) ----
  char* w = (char*)d_ws;
  auto carve = [&](size_t bytes) -> char* {
    char* p = w;
    w += (bytes + 255) & ~(size_t)255;
    return p;
  };
  float* atom     = (float*)carve((size_t)N_ATOMS * 64 * 4);
  float* bond     = (float*)carve((size_t)N_UND * 64 * 4);
  float* angle    = (float*)carve((size_t)N_ANG * 64 * 4);
  float* bw_ag    = (float*)carve((size_t)N_UND * 64 * 4);
  float* bw_bg    = (float*)carve((size_t)N_UND * 64 * 4);
  float* atom_acc = (float*)carve((size_t)N_ATOMS * 64 * 4);
  float* bond_acc = (float*)carve((size_t)N_UND * 64 * 4);
  float* esum     = (float*)carve(2 * N_STRUCT * 4);
  float* cnt      = esum + N_STRUCT;
  bf16_t* p_ac_cw1 = (bf16_t*)carve((size_t)4 * 12288 * 2);
  bf16_t* p_ac_gw1 = (bf16_t*)carve((size_t)4 * 12288 * 2);
  bf16_t* p_ac_cw2 = (bf16_t*)carve((size_t)4 * 4096 * 2);
  bf16_t* p_ac_gw2 = (bf16_t*)carve((size_t)4 * 4096 * 2);
  bf16_t* p_ac_out = (bf16_t*)carve((size_t)4 * 4096 * 2);
  bf16_t* p_bc_cw1 = (bf16_t*)carve((size_t)3 * 16384 * 2);
  bf16_t* p_bc_gw1 = (bf16_t*)carve((size_t)3 * 16384 * 2);
  bf16_t* p_bc_cw2 = (bf16_t*)carve((size_t)3 * 4096 * 2);
  bf16_t* p_bc_gw2 = (bf16_t*)carve((size_t)3 * 4096 * 2);
  bf16_t* p_bc_out = (bf16_t*)carve((size_t)3 * 4096 * 2);
  bf16_t* p_an_cw  = (bf16_t*)carve((size_t)3 * 16384 * 2);
  bf16_t* p_an_gw  = (bf16_t*)carve((size_t)3 * 16384 * 2);
  bf16_t* p_ro_w1  = (bf16_t*)carve((size_t)4096 * 2);
  bf16_t* p_ro_w2  = (bf16_t*)carve((size_t)4096 * 2);

  auto pack = [&](const float* src, bf16_t* dst, int kn) {
    k_pack_w<<<(kn + 255) / 256, 256, 0, stream>>>(src, dst, kn);
  };
  for (int i = 0; i < 4; ++i) {
    pack(ac_cw1 + i * 12288, p_ac_cw1 + i * 12288, 12288);
    pack(ac_gw1 + i * 12288, p_ac_gw1 + i * 12288, 12288);
    pack(ac_cw2 + i * 4096, p_ac_cw2 + i * 4096, 4096);
    pack(ac_gw2 + i * 4096, p_ac_gw2 + i * 4096, 4096);
    pack(ac_out + i * 4096, p_ac_out + i * 4096, 4096);
  }
  for (int i = 0; i < 3; ++i) {
    pack(bc_cw1 + i * 16384, p_bc_cw1 + i * 16384, 16384);
    pack(bc_gw1 + i * 16384, p_bc_gw1 + i * 16384, 16384);
    pack(bc_cw2 + i * 4096, p_bc_cw2 + i * 4096, 4096);
    pack(bc_gw2 + i * 4096, p_bc_gw2 + i * 4096, 4096);
    pack(bc_out + i * 4096, p_bc_out + i * 4096, 4096);
    pack(an_cw + i * 16384, p_an_cw + i * 16384, 16384);
    pack(an_gw + i * 16384, p_an_gw + i * 16384, 16384);
  }
  pack(ro_w1, p_ro_w1, 4096);
  pack(ro_w2, p_ro_w2, 4096);

  // ---- embeddings ----
  k_atom_embed<<<N_ATOMS * 64 / 256, 256, 0, stream>>>(atomic_numbers, emb, atom);
  k_bond_embed<<<N_UND * 64 / 256, 256, 0, stream>>>(bb_ag, bb_bg, Wb, Wbw_ag, Wbw_bg,
                                                     bond, bw_ag, bw_bg);
  k_angle_embed<<<N_ANG * 64 / 256, 256, 0, stream>>>(ang_b, Wa, angle);

  hipMemsetAsync(esum, 0, 2 * N_STRUCT * 4, stream);

  // ---- message-passing iterations ----
  for (int i = 0; i < 4; ++i) {
    hipMemsetAsync(atom_acc, 0, (size_t)N_ATOMS * 64 * 4, stream);
    k_atom_msg<<<N_DIR / 16, 128, 0, stream>>>(
        atom, bond, bw_ag, ag, d2u, p_ac_cw1 + i * 12288, ac_cb1 + i * 64,
        p_ac_gw1 + i * 12288, ac_gb1 + i * 64, p_ac_cw2 + i * 4096, ac_cb2 + i * 64,
        p_ac_gw2 + i * 4096, ac_gb2 + i * 64, atom_acc);
    k_mat_out<<<N_ATOMS / 16, 128, 0, stream>>>(atom_acc, p_ac_out + i * 4096, atom);
    if (i < 3) {
      hipMemsetAsync(bond_acc, 0, (size_t)N_UND * 64 * 4, stream);
      k_bond_msg<<<N_ANG / 16, 128, 0, stream>>>(
          atom, bond, angle, bw_bg, bga, bgb, p_bc_cw1 + i * 16384, bc_cb1 + i * 64,
          p_bc_gw1 + i * 16384, bc_gb1 + i * 64, p_bc_cw2 + i * 4096, bc_cb2 + i * 64,
          p_bc_gw2 + i * 4096, bc_gb2 + i * 64, bond_acc);
      k_mat_out<<<N_UND / 16, 128, 0, stream>>>(bond_acc, p_bc_out + i * 4096, bond);
      k_angle_upd<<<N_ANG / 16, 128, 0, stream>>>(
          atom, bond, angle, bga, bgb, p_an_cw + i * 16384, an_cb + i * 64,
          p_an_gw + i * 16384, an_gb + i * 64);
    }
  }

  // ---- readout ----
  k_readout<<<N_ATOMS / 16, 128, 0, stream>>>(atom, p_ro_w1, ro_b1, p_ro_w2, ro_b2,
                                              ro_w3, ro_b3, owners, esum, cnt);
  k_finalize<<<1, 128, 0, stream>>>(esum, cnt, (float*)d_out);
}
